// VQ_49898930045514
// MI455X (gfx1250) — matrix operand order repaired and checked by reading, not compiled, not run on hardware
//
#include <hip/hip_runtime.h>
#include <stdint.h>

typedef unsigned int       u32;
typedef unsigned long long u64;
typedef unsigned short     u16;

typedef __attribute__((ext_vector_type(16))) __bf16 v16bf;
typedef __attribute__((ext_vector_type(8)))  float  v8f;
typedef __attribute__((ext_vector_type(4)))  u32    v4u;
typedef __attribute__((ext_vector_type(8)))  int    v8i;
typedef __attribute__((ext_vector_type(4)))  int    v4i;

#define NTOK        16384          // 8*2048 tokens
#define D           512            // code dim
#define KS          1024           // split K: [hi(512) | lo(512)]
#define NCODES      8192
#define ROWS_PER_WG 256            // 8 waves * 32 rows (2 row-tiles per wave)
#define NSTRIPS     4
#define STRIP_CODES (NCODES / NSTRIPS)   // 2048
#define CHUNK_CODES 64
#define NCHUNK      (STRIP_CODES / CHUNK_CODES)  // 32
#define ROW_LDS     2064           // 2048B + 2*8B TDM pad per code row
#define CHUNK_BYTES (CHUNK_CODES * ROW_LDS)      // 132096
#define SMEM_BYTES  (2 * CHUNK_BYTES)            // 264192 (<320KB)

#if defined(__has_builtin)
#if __has_builtin(__builtin_amdgcn_tensor_load_to_lds) && \
    __has_builtin(__builtin_amdgcn_s_wait_tensorcnt) && \
    __has_builtin(__builtin_amdgcn_groupstaticsize)
#define USE_TDM 1
#endif
#endif
#ifndef USE_TDM
#define USE_TDM 0
#endif

// ---------- helpers ----------
__device__ __forceinline__ u16 f32_to_bf16_rne(float f) {
    u32 u = __float_as_uint(f);
    u32 r = u + 0x7FFFu + ((u >> 16) & 1u);
    return (u16)(r >> 16);
}
__device__ __forceinline__ float bf16_bits_to_f32(u16 h) {
    return __uint_as_float(((u32)h) << 16);
}

union ABFrag { uint4 q[2]; v16bf v; };

// ---------- prep kernels: bf16 split + code norms ----------
__global__ __launch_bounds__(256) void vq_prep_x(const float* __restrict__ x,
                                                 u16* __restrict__ xs) {
    const int row = blockIdx.x;
    const int t   = threadIdx.x;
#pragma unroll
    for (int j = 0; j < 2; ++j) {
        int   k  = t + j * 256;
        float v  = x[(size_t)row * D + k];
        u16   hi = f32_to_bf16_rne(v);
        float lo = v - bf16_bits_to_f32(hi);
        xs[(size_t)row * KS + k]     = hi;
        xs[(size_t)row * KS + D + k] = f32_to_bf16_rne(lo);
    }
}

__global__ __launch_bounds__(256) void vq_prep_c(const float* __restrict__ cb,
                                                 u16* __restrict__ cs,
                                                 float* __restrict__ c2) {
    __shared__ float red[256];
    const int row = blockIdx.x;
    const int t   = threadIdx.x;
    float s = 0.0f;
#pragma unroll
    for (int j = 0; j < 2; ++j) {
        int   k  = t + j * 256;
        float v  = cb[(size_t)row * D + k];
        u16   hi = f32_to_bf16_rne(v);
        float lo = v - bf16_bits_to_f32(hi);
        cs[(size_t)row * KS + k]     = hi;
        cs[(size_t)row * KS + D + k] = f32_to_bf16_rne(lo);
        s += v * v;
    }
    red[t] = s;
    __syncthreads();
    for (int o = 128; o > 0; o >>= 1) {
        if (t < o) red[t] += red[t + o];
        __syncthreads();
    }
    if (t == 0) c2[row] = red[0];
}

// ---------- TDM issue: 2D tile (64 codes x 2048B) with LDS padding ----------
#if USE_TDM
__device__ __forceinline__ void tdm_issue(const void* gsrc, u32 ldsOff) {
    u64 ga = (u64)(uintptr_t)gsrc;
    v4u g0;
    g0[0] = 1u;                                   // count=1, user mode
    g0[1] = ldsOff;                               // LDS byte address
    g0[2] = (u32)ga;                              // global addr lo
    g0[3] = ((u32)(ga >> 32) & 0x1FFFFFFu) | (2u << 30);  // addr hi | type=2
    v8i g1;
    // data_size=8B(3), pad_enable=1, pad_interval=7(256 DW=1KB), pad_amount=1(2 DW=8B)
    g1[0] = (int)((3u << 16) | (1u << 20) | (7u << 22) | (1u << 25));
    g1[1] = (int)((256u & 0xFFFFu) << 16);        // tensor_dim0 lo16 (256 8B-units/row)
    g1[2] = (int)(((256u >> 16) & 0xFFFFu) | ((8192u & 0xFFFFu) << 16)); // dim0 hi | dim1 lo
    g1[3] = (int)(((8192u >> 16) & 0xFFFFu) | (256u << 16));             // dim1 hi | tile_dim0
    g1[4] = (int)(64u);                           // tile_dim1=64 rows, tile_dim2=0
    g1[5] = (int)(256u);                          // tensor_dim0_stride lo32
    g1[6] = 0;                                    // stride0 hi | stride1 lo
    g1[7] = 0;                                    // stride1 hi
    v4i z4  = {0, 0, 0, 0};
    v8i z8  = {0, 0, 0, 0, 0, 0, 0, 0};
    __builtin_amdgcn_tensor_load_to_lds(g0, g1, z4, z4, z8, 0);
}
#endif

// ---------- main kernel: bf16x3 WMMA distance GEMM + running argmin ----------
// Each wave: 2 row-tiles (32 rows) x 4 col-tiles; B fragments reused across row-tiles.
__global__ __launch_bounds__(256) void vq_gemm_argmin(const u16* __restrict__ xs,
                                                      const u16* __restrict__ cs,
                                                      const float* __restrict__ c2,
                                                      float* __restrict__ stripD,
                                                      int* __restrict__ stripI) {
    extern __shared__ char smem[];
    const int lane  = threadIdx.x & 31;
    const int wave  = threadIdx.x >> 5;
    const int strip = blockIdx.y;
    const int rowBlk = blockIdx.x * 8 + wave;     // 512 row blocks of 32 rows
    const int col  = lane & 15;                   // C-matrix column / A row-in-tile
    const int half = lane >> 4;                   // K-half selector

    const u16* xsRow0 = xs + (size_t)(rowBlk * 32 + col) * KS;
    const u16* xsRow1 = xsRow0 + (size_t)16 * KS;

    float bestD[2][8];
    int   bestI[2][8];
#pragma unroll
    for (int rt = 0; rt < 2; ++rt)
#pragma unroll
        for (int r = 0; r < 8; ++r) { bestD[rt][r] = 3.4e38f; bestI[rt][r] = 0; }

#if USE_TDM
    const u32 ldsBase = __builtin_amdgcn_groupstaticsize();
    if (wave == 0) {
        tdm_issue(cs + (size_t)(strip * STRIP_CODES) * KS, ldsBase);
    }
#endif

    for (int c = 0; c < NCHUNK; ++c) {
#if USE_TDM
        if (wave == 0) {
            if (c + 1 < NCHUNK) {
                tdm_issue(cs + (size_t)(strip * STRIP_CODES + (c + 1) * CHUNK_CODES) * KS,
                          ldsBase + (u32)(((c + 1) & 1) * CHUNK_BYTES));
                __builtin_amdgcn_s_wait_tensorcnt(1);   // chunk c complete (in-order TDM)
            } else {
                __builtin_amdgcn_s_wait_tensorcnt(0);
            }
        }
        __syncthreads();
        const char* Bb = smem + (c & 1) * CHUNK_BYTES;
#else
        __syncthreads();
        {   // cooperative fallback copy (same padded layout)
            const u16* src = cs + (size_t)(strip * STRIP_CODES + c * CHUNK_CODES) * KS;
            for (int u = threadIdx.x; u < CHUNK_CODES * (KS / 8); u += 256) {
                int code  = u >> 7;           // 128 16B units per row
                int unit  = u & 127;
                int kByte = unit * 16;
                uint4 v = *(const uint4*)(src + (size_t)code * KS + unit * 8);
                *(uint4*)(smem + code * ROW_LDS + kByte + ((kByte >> 10) << 3)) = v;
            }
        }
        __syncthreads();
        const char* Bb = smem;
#endif
        v8f acc[2][4];
#pragma unroll
        for (int rt = 0; rt < 2; ++rt)
#pragma unroll
            for (int t = 0; t < 4; ++t) acc[rt][t] = (v8f){0, 0, 0, 0, 0, 0, 0, 0};

        // dot = xh*ch + xh*cl + xl*ch  (bf16x3 split)
        for (int seg = 0; seg < 3; ++seg) {
            const int aOff = (seg == 2) ? 512 : 0;
            const int bOff = (seg == 1) ? 512 : 0;
            for (int kl = 0; kl < 512; kl += 32) {
                // A fragments for both row tiles (batched global loads)
                ABFrag A0, A1;
                const u16* ap0 = xsRow0 + aOff + kl + half * 8;
                const u16* ap1 = xsRow1 + aOff + kl + half * 8;
                A0.q[0] = *(const uint4*)(ap0);
                A0.q[1] = *(const uint4*)(ap0 + 16);
                A1.q[0] = *(const uint4*)(ap1);
                A1.q[1] = *(const uint4*)(ap1 + 16);
                // B fragments for all 4 col tiles into distinct registers (batched ds loads)
                const int kByte = (bOff + kl + half * 8) * 2;
                const int bo    = kByte + ((kByte >> 10) << 3);   // TDM pad adjust
                ABFrag B0, B1, B2, B3;
                {
                    const char* bp0 = Bb + (0 * 16 + col) * ROW_LDS + bo;
                    const char* bp1 = Bb + (1 * 16 + col) * ROW_LDS + bo;
                    const char* bp2 = Bb + (2 * 16 + col) * ROW_LDS + bo;
                    const char* bp3 = Bb + (3 * 16 + col) * ROW_LDS + bo;
                    B0.q[0] = *(const uint4*)(bp0); B0.q[1] = *(const uint4*)(bp0 + 32);
                    B1.q[0] = *(const uint4*)(bp1); B1.q[1] = *(const uint4*)(bp1 + 32);
                    B2.q[0] = *(const uint4*)(bp2); B2.q[1] = *(const uint4*)(bp2 + 32);
                    B3.q[0] = *(const uint4*)(bp3); B3.q[1] = *(const uint4*)(bp3 + 32);
                }
                acc[0][0] = __builtin_amdgcn_wmma_f32_16x16x32_bf16(
                    false, A0.v, false, B0.v, (short)0, acc[0][0], false, false);
                acc[0][1] = __builtin_amdgcn_wmma_f32_16x16x32_bf16(
                    false, A0.v, false, B1.v, (short)0, acc[0][1], false, false);
                acc[0][2] = __builtin_amdgcn_wmma_f32_16x16x32_bf16(
                    false, A0.v, false, B2.v, (short)0, acc[0][2], false, false);
                acc[0][3] = __builtin_amdgcn_wmma_f32_16x16x32_bf16(
                    false, A0.v, false, B3.v, (short)0, acc[0][3], false, false);
                acc[1][0] = __builtin_amdgcn_wmma_f32_16x16x32_bf16(
                    false, A1.v, false, B0.v, (short)0, acc[1][0], false, false);
                acc[1][1] = __builtin_amdgcn_wmma_f32_16x16x32_bf16(
                    false, A1.v, false, B1.v, (short)0, acc[1][1], false, false);
                acc[1][2] = __builtin_amdgcn_wmma_f32_16x16x32_bf16(
                    false, A1.v, false, B2.v, (short)0, acc[1][2], false, false);
                acc[1][3] = __builtin_amdgcn_wmma_f32_16x16x32_bf16(
                    false, A1.v, false, B3.v, (short)0, acc[1][3], false, false);
            }
        }

        // running argmin update: dist = ||c||^2 - 2 x.c  (x^2 constant per row)
        const int codeBase = strip * STRIP_CODES + c * CHUNK_CODES;
#pragma unroll
        for (int t = 0; t < 4; ++t) {
            const int   code = codeBase + t * 16 + col;
            const float cc   = c2[code];
#pragma unroll
            for (int rt = 0; rt < 2; ++rt) {
#pragma unroll
                for (int r = 0; r < 8; ++r) {
                    float dd = cc - 2.0f * acc[rt][t][r];
                    if (dd < bestD[rt][r]) { bestD[rt][r] = dd; bestI[rt][r] = code; }
                }
            }
        }
        __syncthreads();
    }

    // reduce across the 16 lanes holding the same row (columns), keep lowest idx on tie
#pragma unroll
    for (int rt = 0; rt < 2; ++rt) {
#pragma unroll
        for (int r = 0; r < 8; ++r) {
            float d = bestD[rt][r];
            int   i = bestI[rt][r];
#pragma unroll
            for (int off = 1; off < 16; off <<= 1) {
                float od = __shfl_xor(d, off, 32);
                int   oi = __shfl_xor(i, off, 32);
                if (od < d || (od == d && oi < i)) { d = od; i = oi; }
            }
            if (col == 0) {   // lane 0 -> rows 0..7, lane 16 -> rows 8..15
                int row = rowBlk * 32 + rt * 16 + half * 8 + r;
                stripD[(size_t)strip * NTOK + row] = d;
                stripI[(size_t)strip * NTOK + row] = i;
            }
        }
    }
}

// ---------- combine strips, emit indices (as f32) ----------
__global__ __launch_bounds__(256) void vq_combine(const float* __restrict__ sd,
                                                  const int* __restrict__ si,
                                                  int* __restrict__ fi,
                                                  float* __restrict__ outIdx) {
    int r = blockIdx.x * 256 + threadIdx.x;
    float bd = 3.4e38f;
    int   bi = 0;
    for (int s = 0; s < NSTRIPS; ++s) {
        float d = sd[(size_t)s * NTOK + r];
        int   i = si[(size_t)s * NTOK + r];
        if (d < bd || (d == bd && i < bi)) { bd = d; bi = i; }
    }
    fi[r]     = bi;
    outIdx[r] = (float)bi;
}

// ---------- gather codes (exact f32) + per-row loss partials ----------
__global__ __launch_bounds__(256) void vq_gather_loss(const float* __restrict__ cb,
                                                      const float* __restrict__ x,
                                                      const int* __restrict__ fi,
                                                      float* __restrict__ outQ,
                                                      float* __restrict__ rowLoss) {
    __shared__ float red[256];
    const int row = blockIdx.x;
    const int t   = threadIdx.x;
    const int idx = fi[row];
    const float* crow = cb + (size_t)idx * D;
    float s = 0.0f;
#pragma unroll
    for (int j = 0; j < 2; ++j) {
        int   k  = t + j * 256;
        float cv = crow[k];
        float xv = x[(size_t)row * D + k];
        outQ[(size_t)row * D + k] = cv;     // quantized forward value == codes
        float df = xv - cv;
        s += df * df;
    }
    red[t] = s;
    __syncthreads();
    for (int o = 128; o > 0; o >>= 1) {
        if (t < o) red[t] += red[t + o];
        __syncthreads();
    }
    if (t == 0) rowLoss[row] = red[0];
}

__global__ __launch_bounds__(256) void vq_reduce_loss(const float* __restrict__ rl,
                                                      float* __restrict__ outLoss) {
    __shared__ float red[256];
    const int t = threadIdx.x;
    float s = 0.0f;
    for (int j = 0; j < NTOK / 256; ++j) s += rl[(size_t)t + (size_t)j * 256];
    red[t] = s;
    __syncthreads();
    for (int o = 128; o > 0; o >>= 1) {
        if (t < o) red[t] += red[t + o];
        __syncthreads();
    }
    if (t == 0) outLoss[0] = 2.0f * red[0] / (float)(NTOK * D);  // code + commitment loss
}

// ---------- host launch ----------
extern "C" void kernel_launch(void* const* d_in, const int* in_sizes, int n_in,
                              void* d_out, int out_size, void* d_ws, size_t ws_size,
                              hipStream_t stream) {
    const float* x  = (const float*)d_in[0];   // [8,2048,512]
    const float* cb = (const float*)d_in[1];   // [8192,512]

    char* ws = (char*)d_ws;
    u16*   xs       = (u16*)(ws + 0);              // 32 MB
    u16*   cs       = (u16*)(ws + 33554432);       // 16 MB
    float* c2       = (float*)(ws + 50331648);     // 32 KB
    float* stripD   = (float*)(ws + 50364416);     // 256 KB
    int*   stripI   = (int*)(ws + 50626560);       // 256 KB
    int*   finalIdx = (int*)(ws + 50888704);       // 64 KB
    float* rowLoss  = (float*)(ws + 50954240);     // 64 KB

    float* outQ    = (float*)d_out;                // 16384*512
    float* outIdx  = outQ + (size_t)NTOK * D;      // 16384
    float* outLoss = outIdx + NTOK;                // 1

    vq_prep_x<<<NTOK, 256, 0, stream>>>(x, xs);
    vq_prep_c<<<NCODES, 256, 0, stream>>>(cb, cs, c2);

    (void)hipFuncSetAttribute(reinterpret_cast<const void*>(&vq_gemm_argmin),
                              hipFuncAttributeMaxDynamicSharedMemorySize, SMEM_BYTES);
    vq_gemm_argmin<<<dim3(NTOK / ROWS_PER_WG, NSTRIPS), 256, SMEM_BYTES, stream>>>(
        xs, cs, c2, stripD, stripI);

    vq_combine<<<NTOK / 256, 256, 0, stream>>>(stripD, stripI, finalIdx, outIdx);
    vq_gather_loss<<<NTOK, 256, 0, stream>>>(cb, x, finalIdx, outQ, rowLoss);
    vq_reduce_loss<<<1, 256, 0, stream>>>(rowLoss, outLoss);
}
